// FourierCrossAttentionW_7438883356776
// MI455X (gfx1250) — matrix operand order repaired and checked by reading, not compiled, not run on hardware
//
#include <hip/hip_runtime.h>
#include <hip/hip_bf16.h>
#include <math.h>

typedef __attribute__((ext_vector_type(16))) _Float16 v16h;
typedef __attribute__((ext_vector_type(8)))  _Float16 v8h;
typedef __attribute__((ext_vector_type(8)))  float    v8f;

#define BQ 32
#define HH 8
#define EE 64
#define LL 1024
#define TC 64            // time chunk
#define NCHUNK (LL / TC) // 16
#define RS72 72          // padded row stride (halfs) for 64-wide K
#define RS136 136        // padded row stride (halfs) for 128-wide K

struct __align__(16) SharedA {               // stage A temporaries
  _Float16 sQ[EE * RS72];                    // A-operand: [e][t']
  _Float16 sK[EE * RS72];
  _Float16 sFT[128 * RS72];                  // B-operand transposed: [n(Re|Im)][t']
};
struct __align__(16) SharedD {               // stage C/D temporaries
  _Float16 XA[EE * RS136];                   // A-operand: [e][ReX(64)|ImX(64)]
  _Float16 sGT[TC * RS136];                  // B-operand transposed: [t'][k]
};
struct __align__(16) HalfMat { _Float16 re[64 * RS72]; _Float16 im[64 * RS72]; };

struct __align__(16) Shared {
  float ctab[1024];
  float stab[1024];
  union { SharedA a; SharedD d; } u1;        // phase-aliased
  union { HalfMat qftT; HalfMat amat; } u2;  // QftT (stage B A-op) -> Amat (stage C B-op)
  HalfMat kftT;                              // stage B B-operand: [y][e]
  HalfMat kft;                               // stage C A-operand: [e][y]
};

__device__ inline v8f z8() {
  v8f z = {0.f, 0.f, 0.f, 0.f, 0.f, 0.f, 0.f, 0.f};
  return z;
}

// A matrix 16x32 f16, row-major [M][K] in LDS. Lane l: M=m0+(l&15),
// K pairs: klo..klo+7 and klo+16..klo+23 with klo=k0+((l>>4)<<3).
__device__ inline v16h load_A16x32(const _Float16* base, int rs, int m0, int k0, int lane) {
  int m = m0 + (lane & 15);
  int klo = k0 + ((lane >> 4) << 3);
  const _Float16* p = base + m * rs + klo;
  v8h lo = *(const v8h*)(p);
  v8h hi = *(const v8h*)(p + 16);
  v16h r;
#pragma unroll
  for (int i = 0; i < 8; ++i) { r[i] = lo[i]; r[8 + i] = hi[i]; }
  return r;
}

// B matrix 32x16 f16, stored transposed [N][K] in LDS. Lane l: N=n0+(l&15),
// K pairs: kb..kb+7 and kb+16..kb+23 with kb=k0+((l>>4)<<3).
__device__ inline v16h load_B32x16(const _Float16* baseT, int rs, int n0, int k0, int lane) {
  int n = n0 + (lane & 15);
  int kb = k0 + ((lane >> 4) << 3);
  const _Float16* p = baseT + n * rs + kb;
  v8h lo = *(const v8h*)(p);
  v8h hi = *(const v8h*)(p + 16);
  v16h r;
#pragma unroll
  for (int i = 0; i < 8; ++i) { r[i] = lo[i]; r[8 + i] = hi[i]; }
  return r;
}

__device__ inline v8f wmma16(v16h a, v16h b, v8f c) {
  return __builtin_amdgcn_wmma_f32_16x16x32_f16(false, a, false, b, (short)0, c, false, false);
}

// Prefetch one 64x64 (e,t') chunk of q and k into registers (16 floats each).
__device__ inline void load_chunk_regs(const float* __restrict__ q, const float* __restrict__ k,
                                       long baseIO, int t0, int tid,
                                       float fq[16], float fk[16]) {
#pragma unroll
  for (int j = 0; j < 16; ++j) {
    int i = tid + j * 256;
    int e = i & 63;
    int tt = i >> 6;
    long gi = baseIO + (long)(t0 + tt) * (EE * HH) + e * HH;
    fq[j] = q[gi];
    fk[j] = k[gi];
  }
}

__global__ __launch_bounds__(256) void fca_fused_kernel(const float* __restrict__ q,
                                                        const float* __restrict__ k,
                                                        float* __restrict__ out) {
  __shared__ Shared sm;
  const int tid = threadIdx.x;
  const int lane = tid & 31;
  const int w = tid >> 5;        // wave 0..7
  const int bh = blockIdx.x;
  const int b = bh >> 3;
  const int h = bh & 7;

  // One-period twiddle tables: exp(-i*2*pi*j/1024) components.
  const float TWO_PI_N = 6.283185307179586f / 1024.f;
  for (int j = tid; j < 1024; j += 256) {
    float s, c;
    sincosf(TWO_PI_N * (float)j, &s, &c);
    sm.ctab[j] = c;
    sm.stab[j] = s;
  }

  // ---------------- Stage A: Qft/Kft = DFT matmul, K=1024 streamed ----------------
  // Wave w owns output N-tile w of the [64e x 128(Re|Im)] spectra, all 4 M-tiles.
  v8f accQ[4], accK[4];
#pragma unroll
  for (int i = 0; i < 4; ++i) { accQ[i] = z8(); accK[i] = z8(); }

  const long baseIO = (long)b * LL * (EE * HH) + h;  // + (t*64 + e)*8

  float fq[16], fk[16];
  load_chunk_regs(q, k, baseIO, 0, tid, fq, fk);  // prologue prefetch of chunk 0

  for (int ch = 0; ch < NCHUNK; ++ch) {
    const int t0 = ch * TC;
    // Commit the prefetched chunk to LDS (f32 -> f16), [e][t'] layout.
    {
      const int e = tid & 63;
#pragma unroll
      for (int j = 0; j < 16; ++j) {
        int tt = (tid >> 6) + j * 4;
        sm.u1.a.sQ[e * RS72 + tt] = (_Float16)fq[j];
        sm.u1.a.sK[e * RS72 + tt] = (_Float16)fk[j];
      }
    }
    // DFT twiddles for this chunk: row n<64 -> cos, n>=64 -> -sin (rfft sign).
    for (int i = tid; i < 128 * TC; i += 256) {
      int n = i >> 6;
      int tt = i & 63;
      int m = n & 63;
      int idx = (m * (t0 + tt)) & 1023;
      float v = (n < 64) ? sm.ctab[idx] : -sm.stab[idx];
      sm.u1.a.sFT[n * RS72 + tt] = (_Float16)v;
    }
    __syncthreads();

    // Kick off global prefetch of the next chunk; HBM latency hides under WMMAs.
    if (ch + 1 < NCHUNK)
      load_chunk_regs(q, k, baseIO, t0 + TC, tid, fq, fk);

#pragma unroll
    for (int kk = 0; kk < TC; kk += 32) {
      v16h bop = load_B32x16(sm.u1.a.sFT, RS72, w * 16, kk, lane);
      v16h aq[4], ak[4];
#pragma unroll
      for (int mt = 0; mt < 4; ++mt) aq[mt] = load_A16x32(sm.u1.a.sQ, RS72, mt * 16, kk, lane);
#pragma unroll
      for (int mt = 0; mt < 4; ++mt) accQ[mt] = wmma16(aq[mt], bop, accQ[mt]);
#pragma unroll
      for (int mt = 0; mt < 4; ++mt) ak[mt] = load_A16x32(sm.u1.a.sK, RS72, mt * 16, kk, lane);
#pragma unroll
      for (int mt = 0; mt < 4; ++mt) accK[mt] = wmma16(ak[mt], bop, accK[mt]);
    }
    __syncthreads();
  }

  // Scatter spectra to LDS in the layouts later stages need.
  {
    const int cn = w * 16 + (lane & 15);     // spectrum column (0..127)
    const int rowadd = (lane >> 4) << 3;     // D-tile: lanes 16..31 hold M+8
#pragma unroll
    for (int mt = 0; mt < 4; ++mt) {
#pragma unroll
      for (int r = 0; r < 8; ++r) {
        int e = mt * 16 + r + rowadd;
        float vq = accQ[mt][r];
        float vk = accK[mt][r];
        if (cn < 64) {
          sm.u2.qftT.re[cn * RS72 + e] = (_Float16)vq;  // [x][e] (stage B A-op)
          sm.kftT.re[cn * RS72 + e] = (_Float16)vk;     // [y][e] (stage B B-op)
          sm.kft.re[e * RS72 + cn] = (_Float16)vk;      // [e][y] (stage C A-op)
        } else {
          int m = cn - 64;
          sm.u2.qftT.im[m * RS72 + e] = (_Float16)vq;
          sm.kftT.im[m * RS72 + e] = (_Float16)vk;
          sm.kft.im[e * RS72 + m] = (_Float16)vk;
        }
      }
    }
  }
  __syncthreads();

  // ---------------- Stage B: A[x,y] = tanh(sum_e Qft[e,x]*Kft[e,y]) (complex) ----------------
  // waves 0..3 -> Re part (N-tile = w), waves 4..7 -> Im part (N-tile = w-4).
  {
    const int part = w >> 2;
    const int nt = w & 3;
    v8f res[4];
#pragma unroll
    for (int mt = 0; mt < 4; ++mt) {
      v8f a1 = z8(), a2 = z8();
#pragma unroll
      for (int kk = 0; kk < 64; kk += 32) {
        v16h A_re = load_A16x32(sm.u2.qftT.re, RS72, mt * 16, kk, lane);
        v16h A_im = load_A16x32(sm.u2.qftT.im, RS72, mt * 16, kk, lane);
        v16h B_re = load_B32x16(sm.kftT.re, RS72, nt * 16, kk, lane);
        v16h B_im = load_B32x16(sm.kftT.im, RS72, nt * 16, kk, lane);
        if (part == 0) { a1 = wmma16(A_re, B_re, a1); a2 = wmma16(A_im, B_im, a2); }
        else           { a1 = wmma16(A_re, B_im, a1); a2 = wmma16(A_im, B_re, a2); }
      }
      v8f d = (part == 0) ? (a1 - a2) : (a1 + a2);
#pragma unroll
      for (int r = 0; r < 8; ++r) d[r] = tanhf(d[r]);
      res[mt] = d;
    }
    __syncthreads();  // qftT reads done before amat overwrites the same LDS
    const int cy = nt * 16 + (lane & 15);
    const int rowadd = (lane >> 4) << 3;
    _Float16* dst = (part == 0) ? sm.u2.amat.re : sm.u2.amat.im;
#pragma unroll
    for (int mt = 0; mt < 4; ++mt)
#pragma unroll
      for (int r = 0; r < 8; ++r) {
        int x = mt * 16 + r + rowadd;
        dst[x * RS72 + cy] = (_Float16)res[mt][r];  // Amat row-major == B-op transposed layout
      }
  }
  __syncthreads();

  // ---------------- Stage C: X[e,x] = sum_y Kft[e,y]*A[x,y] (complex) ----------------
  {
    const int part = w >> 2;
    const int xt = w & 3;
    v8f res[4];
#pragma unroll
    for (int et = 0; et < 4; ++et) {
      v8f a1 = z8(), a2 = z8();
#pragma unroll
      for (int kk = 0; kk < 64; kk += 32) {
        v16h A_re = load_A16x32(sm.kft.re, RS72, et * 16, kk, lane);
        v16h A_im = load_A16x32(sm.kft.im, RS72, et * 16, kk, lane);
        v16h B_re = load_B32x16(sm.u2.amat.re, RS72, xt * 16, kk, lane);
        v16h B_im = load_B32x16(sm.u2.amat.im, RS72, xt * 16, kk, lane);
        if (part == 0) { a1 = wmma16(A_re, B_re, a1); a2 = wmma16(A_im, B_im, a2); }
        else           { a1 = wmma16(A_re, B_im, a1); a2 = wmma16(A_im, B_re, a2); }
      }
      res[et] = (part == 0) ? (a1 - a2) : (a1 + a2);
    }
    __syncthreads();
    const int cx = xt * 16 + (lane & 15);
    const int rowadd = (lane >> 4) << 3;
    const int coff = (part == 0) ? 0 : 64;
#pragma unroll
    for (int et = 0; et < 4; ++et)
#pragma unroll
      for (int r = 0; r < 8; ++r) {
        int e = et * 16 + r + rowadd;
        sm.u1.d.XA[e * RS136 + coff + cx] = (_Float16)res[et][r];
      }
  }
  __syncthreads();

  // ---------------- Stage D: irfft matmul, stream t in chunks ----------------
  // out[e,t] = sum_m s_m*(ReX*cos - ImX*sin), s_0=1 (Im dropped for m=0), s_m=2.
  const float SCALE = 1.0f / (1024.0f * 64.0f * 64.0f);  // 1/N * 1/IN_CH * 1/OUT_CH
  for (int ch = 0; ch < NCHUNK; ++ch) {
    const int t0 = ch * TC;
    for (int i = tid; i < TC * 128; i += 256) {
      int tt = i >> 7;
      int kx = i & 127;
      int m = kx & 63;
      int idx = (m * (t0 + tt)) & 1023;
      float v;
      if (kx < 64) v = (m == 0 ? 1.0f : 2.0f) * sm.ctab[idx];
      else         v = (m == 0 ? 0.0f : -2.0f * sm.stab[idx]);
      sm.u1.d.sGT[tt * RS136 + kx] = (_Float16)v;
    }
    __syncthreads();
    const int nt = w & 3;
#pragma unroll
    for (int half = 0; half < 2; ++half) {
      const int mt = (w >> 2) + half * 2;
      v8f acc = z8();
      v16h Aop[4], Bop[4];
#pragma unroll
      for (int kk = 0; kk < 128; kk += 32) {
        Aop[kk >> 5] = load_A16x32(sm.u1.d.XA, RS136, mt * 16, kk, lane);
        Bop[kk >> 5] = load_B32x16(sm.u1.d.sGT, RS136, nt * 16, kk, lane);
      }
#pragma unroll
      for (int kk = 0; kk < 128; kk += 32) acc = wmma16(Aop[kk >> 5], Bop[kk >> 5], acc);
      const int ct = t0 + nt * 16 + (lane & 15);
      const int rowadd = (lane >> 4) << 3;
#pragma unroll
      for (int r = 0; r < 8; ++r) {
        int e = mt * 16 + r + rowadd;
        out[baseIO + (long)ct * (EE * HH) + e * HH] = acc[r] * SCALE;
      }
    }
    __syncthreads();  // protect sGT refill next chunk
  }
}

extern "C" void kernel_launch(void* const* d_in, const int* in_sizes, int n_in,
                              void* d_out, int out_size, void* d_ws, size_t ws_size,
                              hipStream_t stream) {
  const float* q = (const float*)d_in[0];
  const float* k = (const float*)d_in[1];
  // d_in[2] (v) and d_in[3] (mask) are unused by the reference forward.
  float* out = (float*)d_out;
  dim3 grid(BQ * HH);   // one workgroup per (b,h)
  fca_fused_kernel<<<grid, 256, 0, stream>>>(q, k, out);
}
